// TransformerEncoderLayer_43456479101333
// MI455X (gfx1250) — matrix operand order repaired and checked by reading, not compiled, not run on hardware
//
#include <hip/hip_runtime.h>
#include <hip/hip_bf16.h>
#include <math.h>

// ===========================================================================
// MI455X (gfx1250) TransformerEncoderLayer, wave32 + WMMA f32_16x16x32_f16.
// Bandwidth floor: bias tensor 128MB read once (flash-style attention, no
// S x S materialization). All GEMMs on the WMMA pipe, fp32 accumulation.
// GEMM staging uses gfx1250 async global->LDS copies (ASYNCcnt) with double
// buffering when the toolchain exposes the builtins; sync b128 path otherwise.
// ===========================================================================

typedef __attribute__((ext_vector_type(16))) _Float16 v16h;
typedef __attribute__((ext_vector_type(8)))  float    v8f;
typedef __attribute__((ext_vector_type(4)))  int      v4i;

union Frag { v16h h; unsigned int u[8]; };

constexpr int BB = 4, SS = 1024, EE = 512, HH = 8, DK = 64, FF = 2048;
constexpr int ROWS = BB * SS;   // 4096

#if defined(__has_builtin)
#if __has_builtin(__builtin_amdgcn_global_load_async_to_lds_b128) && \
    __has_builtin(__builtin_amdgcn_s_wait_asynccnt)
#define HAVE_ASYNC_LDS 1
#endif
#endif
#ifndef HAVE_ASYNC_LDS
#define HAVE_ASYNC_LDS 0
#endif

#if HAVE_ASYNC_LDS
typedef __attribute__((address_space(1))) v4i gv4i;   // global
typedef __attribute__((address_space(3))) v4i sv4i;   // LDS
#endif

// 16-byte global -> LDS copy: async DMA path (no VGPR bounce, ASYNCcnt) or
// synchronous vector load/store fallback.
__device__ inline void copy16_g2l(const _Float16* gsrc, _Float16* ldst) {
#if HAVE_ASYNC_LDS
  // AS1: generic address == global address.  AS3: generic LDS pointer's low
  // 32 bits are the LDS byte offset (ISA 10.2 aperture rule), so the integer
  // round-trip is exact.
  gv4i* g = (gv4i*)(unsigned long long)gsrc;
  sv4i* l = (sv4i*)(unsigned int)(unsigned long long)ldst;
  __builtin_amdgcn_global_load_async_to_lds_b128(g, l, 0, 0);
#else
  *(uint4*)ldst = *(const uint4*)gsrc;
#endif
}

__device__ inline void wait_stage_done() {
#if HAVE_ASYNC_LDS
  __builtin_amdgcn_s_wait_asynccnt(0);
#endif
}

// ---- WMMA fragment loaders (layouts per CDNA5 ISA 7.12.2) ------
// A (16x32 f16): lane&15 = M row; halves -> K = pat(v) + 8*(lane>>4)
__device__ inline void load_fragA(const _Float16* base, int stride, Frag& f) {
  const int lane = threadIdx.x & 31;
  const int lhi  = lane >> 4;
  const _Float16* p = base + (size_t)(lane & 15) * stride + 8 * lhi;
#pragma unroll
  for (int v = 0; v < 8; ++v) {
    const int k = (v < 4) ? (2 * v) : (16 + 2 * (v - 4));
    f.u[v] = *(const unsigned int*)(p + k);
  }
}
// B (32x16 f16): lane&15 = N col; halves -> K = pat(v) + 8*(lane>>4)
// base must point at [N][K] storage (K contiguous per column).
__device__ inline void load_fragB(const _Float16* base, int stride, Frag& f) {
  const int lane = threadIdx.x & 31;
  const int lhi  = lane >> 4;
  const _Float16* p = base + (size_t)(lane & 15) * stride + 8 * lhi;
#pragma unroll
  for (int v = 0; v < 8; ++v) f.u[v] = *(const unsigned int*)(p + ((v < 4) ? (2 * v) : (16 + 2 * (v - 4))));
}

__device__ inline v8f wmma16x16x32(const Frag& a, const Frag& b, v8f c) {
  return __builtin_amdgcn_wmma_f32_16x16x32_f16(false, a.h, false, b.h,
                                                (short)0, c, false, false);
}

// ---- conversion / transpose helpers ---------------------------------------
__global__ __launch_bounds__(256)
void cvt_f32_to_f16_kernel(const float* __restrict__ s, _Float16* __restrict__ d, int n) {
  const int i = blockIdx.x * 256 + threadIdx.x;
  if (i < n) d[i] = (_Float16)s[i];
}

// src f32 [rows][cols] -> dst f16 [cols][rows]  (weight pre-transpose)
__global__ __launch_bounds__(256)
void transpose_to_f16_kernel(const float* __restrict__ s, _Float16* __restrict__ d,
                             int rows, int cols) {
  const int i = blockIdx.x * 256 + threadIdx.x;
  if (i < rows * cols) {
    const int r = i / cols, c = i - r * cols;
    d[(size_t)c * rows + r] = (_Float16)s[i];
  }
}

// ---- generic WMMA GEMM: C[M,N] = A[M,K] (f16) @ Bt[N,K]^T (f16) + bias ----
enum { EPI_F16 = 0, EPI_F32 = 1, EPI_GELU_F16 = 2, EPI_QHEAD = 3, EPI_KVHEAD = 4 };

// stage one 128x32 A tile + 64x32 B tile into LDS (3 x 16B per thread)
__device__ inline void stage_tiles(const _Float16* __restrict__ A,
                                   const _Float16* __restrict__ Bt,
                                   _Float16* la, _Float16* lb,
                                   int m0, int n0, int K, int kk, int tid) {
#pragma unroll
  for (int c = 0; c < 2; ++c) {
    const int idx = tid + c * 256;
    const int row = idx >> 2, kc = (idx & 3) * 8;
    copy16_g2l(&A[(size_t)(m0 + row) * K + kk + kc], &la[row * 40 + kc]);
  }
  {
    const int row = tid >> 2, kc = (tid & 3) * 8;
    copy16_g2l(&Bt[(size_t)(n0 + row) * K + kk + kc], &lb[row * 40 + kc]);
  }
}

__global__ __launch_bounds__(256)
void wmma_gemm_kernel(const _Float16* __restrict__ A,
                      const _Float16* __restrict__ Bt,
                      const float* __restrict__ bias,
                      void* __restrict__ out0, void* __restrict__ out1,
                      int M, int N, int K, int mode)
{
  // double-buffered tiles: 2 * (10240 + 5120) B = 30 KB of the 320 KB WGP LDS
  __shared__ __align__(16) _Float16 lA[2][128 * 40];   // 128 x 32, pad 8
  __shared__ __align__(16) _Float16 lB[2][64 * 40];    //  64 x 32 (N-major)

  const int tid  = threadIdx.x;
  const int wave = tid >> 5;
  const int m0 = blockIdx.y * 128;
  const int n0 = blockIdx.x * 64;
  const int wm = (wave >> 1) * 32;
  const int wn = (wave & 1) * 32;

  v8f z = {};
  v8f acc00 = z, acc01 = z, acc10 = z, acc11 = z;

  stage_tiles(A, Bt, lA[0], lB[0], m0, n0, K, 0, tid);   // prologue
  int buf = 0;
  for (int kk = 0; kk < K; kk += 32) {
    wait_stage_done();         // our async writes for lA/lB[buf] are in LDS
    __syncthreads();           // everyone's are; also fences reads of buf^1
    if (kk + 64 < K)
      __builtin_prefetch(&A[(size_t)(m0 + (tid >> 1)) * K + kk + 64], 0, 1);
    if (kk + 32 < K)           // overlap next tile's DMA with this tile's math
      stage_tiles(A, Bt, lA[buf ^ 1], lB[buf ^ 1], m0, n0, K, kk + 32, tid);

    Frag a0, a1, b0, b1;
    load_fragA(&lA[buf][(wm +  0) * 40], 40, a0);
    load_fragA(&lA[buf][(wm + 16) * 40], 40, a1);
    load_fragB(&lB[buf][(wn +  0) * 40], 40, b0);
    load_fragB(&lB[buf][(wn + 16) * 40], 40, b1);
    acc00 = wmma16x16x32(a0, b0, acc00);
    acc01 = wmma16x16x32(a0, b1, acc01);
    acc10 = wmma16x16x32(a1, b0, acc10);
    acc11 = wmma16x16x32(a1, b1, acc11);
    buf ^= 1;
  }

  const int lane = tid & 31, lhi = lane >> 4, ln = lane & 15;
  v8f accs[2][2] = {{acc00, acc01}, {acc10, acc11}};
#pragma unroll
  for (int i = 0; i < 2; ++i)
#pragma unroll
    for (int j = 0; j < 2; ++j) {
      const int gn = n0 + wn + 16 * j + ln;
      const float bv = bias[gn];
#pragma unroll
      for (int v = 0; v < 8; ++v) {
        const int gm = m0 + wm + 16 * i + v + 8 * lhi;   // C layout: VGPR v -> M
        float val = accs[i][j][v] + bv;
        if (mode == EPI_F16) {
          ((_Float16*)out0)[(size_t)gm * N + gn] = (_Float16)val;
        } else if (mode == EPI_F32) {
          ((float*)out0)[(size_t)gm * N + gn] = val;
        } else if (mode == EPI_GELU_F16) {
          val = 0.5f * val * (1.0f + erff(val * 0.70710678118654752f));
          ((_Float16*)out0)[(size_t)gm * N + gn] = (_Float16)val;
        } else {   // head-split layouts for attention
          const int b = gm >> 10, s = gm & (SS - 1);
          const int h = gn >> 6,  d = gn & (DK - 1);
          const size_t bh = (size_t)(b * HH + h);
          ((_Float16*)out0)[(bh * SS + s) * DK + d] = (_Float16)val;
          if (mode == EPI_KVHEAD)   // transposed copy [bh][dk][s] for P@V B-frags
            ((_Float16*)out1)[(bh * DK + d) * SS + s] = (_Float16)val;
        }
      }
    }
}

// ---- flash-style attention: softmax(q k^T/8 + bias) @ v -------------------
__global__ __launch_bounds__(256)
void attn_flash_kernel(const _Float16* __restrict__ qh,   // [bh][s][dk]
                       const _Float16* __restrict__ kvh,  // [bh][s][dk]
                       const _Float16* __restrict__ kvT,  // [bh][dk][s]
                       const float* __restrict__ bias,    // [b][h][s][s]
                       _Float16* __restrict__ ctxh)       // [b][s][e]
{
  __shared__ __align__(16) _Float16 ldsP[8 * 16 * 32];    // per-wave P staging
  const int tid = threadIdx.x, wave = tid >> 5;
  const int lane = tid & 31, lhi = lane >> 4, ln = lane & 15;
  const int b = blockIdx.z, h = blockIdx.y, bh = b * HH + h;
  const int q0 = blockIdx.x * 128 + wave * 16;

  Frag qf0, qf1;
  const _Float16* qbase = qh + ((size_t)bh * SS + q0) * DK;
  load_fragA(qbase +  0, DK, qf0);                // dk 0..31
  load_fragA(qbase + 32, DK, qf1);                // dk 32..63

  v8f z = {};
  v8f acc[4]; acc[0] = z; acc[1] = z; acc[2] = z; acc[3] = z;
  float mrun[8], lrun[8];
#pragma unroll
  for (int v = 0; v < 8; ++v) { mrun[v] = -1e30f; lrun[v] = 0.f; }

  _Float16* pbuf = &ldsP[wave * 16 * 32];

  for (int kb = 0; kb < SS / 32; ++kb) {          // 32 keys per iteration
    v8f sc0 = z, sc1 = z;
#pragma unroll
    for (int t = 0; t < 2; ++t) {
      const _Float16* kbase = kvh + ((size_t)bh * SS + (kb * 2 + t) * 16) * DK;
      Frag kf0, kf1;
      load_fragB(kbase +  0, DK, kf0);
      load_fragB(kbase + 32, DK, kf1);
      v8f s = (t == 0) ? sc0 : sc1;
      s = wmma16x16x32(qf0, kf0, s);
      s = wmma16x16x32(qf1, kf1, s);
      if (t == 0) sc0 = s; else sc1 = s;
    }
    float s0[8], s1[8];
#pragma unroll
    for (int v = 0; v < 8; ++v) {                 // scale + bias (read once)
      const int qrow = q0 + v + 8 * lhi;
      const float* bp = bias + ((size_t)bh * SS + qrow) * SS + kb * 32;
      s0[v] = sc0[v] * 0.125f + bp[ln];
      s1[v] = sc1[v] * 0.125f + bp[16 + ln];
    }
#pragma unroll
    for (int v = 0; v < 8; ++v) {                 // online softmax per row
      float mt = fmaxf(s0[v], s1[v]);
      for (int m = 1; m < 16; m <<= 1) mt = fmaxf(mt, __shfl_xor(mt, m, 32));
      const float mnew = fmaxf(mrun[v], mt);
      const float corr = __expf(mrun[v] - mnew);
      const float p0 = __expf(s0[v] - mnew);
      const float p1 = __expf(s1[v] - mnew);
      float ps = p0 + p1;
      for (int m = 1; m < 16; m <<= 1) ps += __shfl_xor(ps, m, 32);
      lrun[v] = lrun[v] * corr + ps;
      mrun[v] = mnew;
#pragma unroll
      for (int j = 0; j < 4; ++j) acc[j][v] *= corr;
      const int r = v + 8 * lhi;                  // transpose P into A layout
      pbuf[r * 32 + ln]      = (_Float16)p0;
      pbuf[r * 32 + 16 + ln] = (_Float16)p1;
    }
    __syncthreads();
    Frag pf; load_fragA(pbuf, 32, pf);
#pragma unroll
    for (int j = 0; j < 4; ++j) {                 // ctx += P @ V
      Frag vf;
      load_fragB(kvT + ((size_t)bh * DK + 16 * j) * SS + kb * 32, SS, vf);
      acc[j] = wmma16x16x32(pf, vf, acc[j]);
    }
    __syncthreads();
  }
#pragma unroll
  for (int v = 0; v < 8; ++v) {
    const float inv = 1.0f / lrun[v];
    const int row = q0 + v + 8 * lhi;
    _Float16* op = ctxh + ((size_t)b * SS + row) * EE + h * DK;
#pragma unroll
    for (int j = 0; j < 4; ++j) op[16 * j + ln] = (_Float16)(acc[j][v] * inv);
  }
}

// ---- residual + LayerNorm (one 256-thread block per row of E=512) ---------
__global__ __launch_bounds__(256)
void add_layernorm_kernel(const float* __restrict__ xres,
                          const float* __restrict__ y,
                          const float* __restrict__ gain,
                          const float* __restrict__ beta,
                          float* __restrict__ out,
                          _Float16* __restrict__ out_h)
{
  __shared__ float r1[256], r2[256];
  const int row = blockIdx.x, tid = threadIdx.x;
  const float* yr = y + (size_t)row * EE;
  const float v0 = yr[tid], v1 = yr[tid + 256];
  r1[tid] = v0 + v1;
  r2[tid] = v0 * v0 + v1 * v1;
  __syncthreads();
  for (int off = 128; off > 0; off >>= 1) {
    if (tid < off) { r1[tid] += r1[tid + off]; r2[tid] += r2[tid + off]; }
    __syncthreads();
  }
  const float mu   = r1[0] * (1.0f / EE);
  const float var  = r2[0] * (1.0f / EE) - mu * mu;
  const float rstd = rsqrtf(var + 1e-5f);
  const float* xr = xres + (size_t)row * EE;
#pragma unroll
  for (int c = 0; c < 2; ++c) {
    const int i = tid + c * 256;
    const float o = xr[i] + (yr[i] - mu) * rstd * gain[i] + beta[i];
    out[(size_t)row * EE + i] = o;
    if (out_h) out_h[(size_t)row * EE + i] = (_Float16)o;
  }
}

// ===========================================================================
extern "C" void kernel_launch(void* const* d_in, const int* in_sizes, int n_in,
                              void* d_out, int out_size, void* d_ws, size_t ws_size,
                              hipStream_t stream) {
  (void)in_sizes; (void)n_in; (void)out_size; (void)ws_size;
  const float* x   = (const float*)d_in[0];
  const float* bias= (const float*)d_in[1];
  const float* Wq  = (const float*)d_in[2];  const float* bq = (const float*)d_in[3];
  const float* Wv  = (const float*)d_in[4];  const float* bv = (const float*)d_in[5];
  const float* Wo  = (const float*)d_in[6];  const float* bo = (const float*)d_in[7];
  const float* W1  = (const float*)d_in[8];  const float* b1 = (const float*)d_in[9];
  const float* W2  = (const float*)d_in[10]; const float* b2 = (const float*)d_in[11];
  const float* g1  = (const float*)d_in[12]; const float* lb1= (const float*)d_in[13];
  const float* g2  = (const float*)d_in[14]; const float* lb2= (const float*)d_in[15];
  float* out = (float*)d_out;

  // workspace carve-up (256B aligned), total ~63 MB
  char* w = (char*)d_ws; size_t off = 0;
  auto alloc = [&](size_t bytes) -> void* {
    void* p = w + off; off += (bytes + 255) & ~(size_t)255; return p;
  };
  _Float16* x_h  = (_Float16*)alloc((size_t)ROWS * EE * 2);
  _Float16* Wq_t = (_Float16*)alloc((size_t)EE * EE * 2);
  _Float16* Wv_t = (_Float16*)alloc((size_t)EE * EE * 2);
  _Float16* Wo_t = (_Float16*)alloc((size_t)EE * EE * 2);
  _Float16* W1_t = (_Float16*)alloc((size_t)EE * FF * 2);
  _Float16* W2_t = (_Float16*)alloc((size_t)FF * EE * 2);
  _Float16* q_h  = (_Float16*)alloc((size_t)ROWS * EE * 2);
  _Float16* kv_h = (_Float16*)alloc((size_t)ROWS * EE * 2);
  _Float16* kvT_h= (_Float16*)alloc((size_t)ROWS * EE * 2);
  _Float16* ctx_h= (_Float16*)alloc((size_t)ROWS * EE * 2);
  float*    ybuf = (float*)   alloc((size_t)ROWS * EE * 4);
  float*    x1   = (float*)   alloc((size_t)ROWS * EE * 4);
  _Float16* x1_h = (_Float16*)alloc((size_t)ROWS * EE * 2);
  _Float16* h1_h = (_Float16*)alloc((size_t)ROWS * FF * 2);

  const dim3 blk(256);
  // 0) precision staging: x -> f16, weights -> f16 transposed [N][K]
  cvt_f32_to_f16_kernel<<<(ROWS * EE + 255) / 256, blk, 0, stream>>>(x, x_h, ROWS * EE);
  transpose_to_f16_kernel<<<(EE * EE + 255) / 256, blk, 0, stream>>>(Wq, Wq_t, EE, EE);
  transpose_to_f16_kernel<<<(EE * EE + 255) / 256, blk, 0, stream>>>(Wv, Wv_t, EE, EE);
  transpose_to_f16_kernel<<<(EE * EE + 255) / 256, blk, 0, stream>>>(Wo, Wo_t, EE, EE);
  transpose_to_f16_kernel<<<(EE * FF + 255) / 256, blk, 0, stream>>>(W1, W1_t, EE, FF);
  transpose_to_f16_kernel<<<(FF * EE + 255) / 256, blk, 0, stream>>>(W2, W2_t, FF, EE);

  // 1) q = x@Wq+bq ; kv = x@Wv+bv (reference uses Wv for both k and v)
  wmma_gemm_kernel<<<dim3(EE / 64, ROWS / 128), blk, 0, stream>>>(
      x_h, Wq_t, bq, q_h, nullptr, ROWS, EE, EE, EPI_QHEAD);
  wmma_gemm_kernel<<<dim3(EE / 64, ROWS / 128), blk, 0, stream>>>(
      x_h, Wv_t, bv, kv_h, kvT_h, ROWS, EE, EE, EPI_KVHEAD);

  // 2) streaming-softmax attention (bias fused, read exactly once)
  attn_flash_kernel<<<dim3(SS / 128, HH, BB), blk, 0, stream>>>(
      q_h, kv_h, kvT_h, bias, ctx_h);

  // 3) y = ctx@Wo+bo ; x1 = x + LN(y)
  wmma_gemm_kernel<<<dim3(EE / 64, ROWS / 128), blk, 0, stream>>>(
      ctx_h, Wo_t, bo, ybuf, nullptr, ROWS, EE, EE, EPI_F32);
  add_layernorm_kernel<<<ROWS, blk, 0, stream>>>(x, ybuf, g1, lb1, x1, x1_h);

  // 4) FFN: h1 = gelu(x1@W1+b1) ; y2 = h1@W2+b2 ; out = x1 + LN(y2)
  wmma_gemm_kernel<<<dim3(FF / 64, ROWS / 128), blk, 0, stream>>>(
      x1_h, W1_t, b1, h1_h, nullptr, ROWS, FF, EE, EPI_GELU_F16);
  wmma_gemm_kernel<<<dim3(EE / 64, ROWS / 128), blk, 0, stream>>>(
      h1_h, W2_t, b2, ybuf, nullptr, ROWS, EE, FF, EPI_F32);
  add_layernorm_kernel<<<ROWS, blk, 0, stream>>>(x1, ybuf, g2, lb2, out, nullptr);
}
